// PAM_5626407157850
// MI455X (gfx1250) — matrix operand order repaired and checked by reading, not compiled, not run on hardware
//
#include <hip/hip_runtime.h>
#include <hip/hip_bf16.h>
#include <stdint.h>

// Sizes from reference: B=4, H=W=64, C=256, MID=256, N=4096, 3C=768, C/8=32
#define NBATCH 4
#define NN     4096      // H*W
#define CC     256
#define C3     768
#define QK     32        // C/8
#define QKVW   320       // 32 (Q) + 32 (K) + 256 (V)
#define NPIX   16384     // B*H*W

typedef __attribute__((ext_vector_type(16))) __bf16   v16bf;
typedef __attribute__((ext_vector_type(8)))  float    v8f;
typedef __attribute__((ext_vector_type(4)))  unsigned v4u;
typedef __attribute__((ext_vector_type(2)))  unsigned v2u;

union V16U { v16bf v; v4u q[2]; };

__device__ __forceinline__ unsigned short f2bf(float f) {
    unsigned u = __float_as_uint(f);
    u += 0x7FFFu + ((u >> 16) & 1u);          // round-to-nearest-even
    return (unsigned short)(u >> 16);
}

// ---------------------------------------------------------------------------
// Kernel 1: fold w_proj into q/k/v weights: Wqkv[k][col] (768 x 320), bf16.
//   col 0..31  -> w_proj @ w_b ; 32..63 -> w_proj @ w_c ; 64..319 -> w_proj @ w_d
// ---------------------------------------------------------------------------
__global__ void pam_fuse_weights(const float* __restrict__ wp,
                                 const float* __restrict__ wb,
                                 const float* __restrict__ wc,
                                 const float* __restrict__ wd,
                                 unsigned short* __restrict__ wqkv) {
    int t = blockIdx.x * 256 + threadIdx.x;
    if (t >= C3 * QKVW) return;
    int k = t / QKVW, col = t - k * QKVW;
    const float* wrow = wp + (size_t)k * 256;
    float s = 0.f;
    if (col < 32)       { for (int i = 0; i < 256; ++i) s += wrow[i] * wb[i * 32 + col]; }
    else if (col < 64)  { int c = col - 32; for (int i = 0; i < 256; ++i) s += wrow[i] * wc[i * 32 + c]; }
    else                { int c = col - 64; for (int i = 0; i < 256; ++i) s += wrow[i] * wd[i * 256 + c]; }
    wqkv[t] = f2bf(s);
}

// ---------------------------------------------------------------------------
// Kernel 2: depthwise 5x5 SRM conv, depth_multiplier=3, SAME padding.
//   f[p][j] = sum_{5x5} x[b, y+dy-2, x+dx-2, j/3] * w_dw[dy][dx][0][j]  (bf16 out)
// ---------------------------------------------------------------------------
__global__ void pam_dwconv(const float* __restrict__ x,
                           const float* __restrict__ wdw,
                           unsigned short* __restrict__ f) {
    int t = blockIdx.x * 256 + threadIdx.x;
    if (t >= NPIX * C3) return;
    int p = t / C3, j = t - p * C3;
    int b = p >> 12, y = (p >> 6) & 63, xw = p & 63;
    int cin = j / 3;
    float s = 0.f;
    #pragma unroll
    for (int dy = 0; dy < 5; ++dy) {
        int yy = y + dy - 2;
        if ((unsigned)yy >= 64u) continue;
        #pragma unroll
        for (int dx = 0; dx < 5; ++dx) {
            int xx = xw + dx - 2;
            if ((unsigned)xx >= 64u) continue;
            s += x[(((size_t)(b * 64 + yy) * 64 + xx) * 256) + cin] * wdw[(dy * 5 + dx) * C3 + j];
        }
    }
    f[(size_t)p * C3 + j] = f2bf(s);
}

// ---------------------------------------------------------------------------
// Kernel 3: QKV GEMM, (16384 x 768) @ (768 x 320) -> qkv bf16, via WMMA bf16.
// 8 waves/WG, each wave owns a 16-row stripe x all 20 column tiles.
// Wqkv k-chunks staged into LDS transposed (pitch 40); B operands double-
// buffered so ds_load latency overlaps the WMMA pipe (partial dscnt waits).
// ---------------------------------------------------------------------------
__global__ __launch_bounds__(256) void pam_qkv_gemm(const unsigned short* __restrict__ f,
                                                    const unsigned short* __restrict__ wqkv,
                                                    unsigned short* __restrict__ qkv) {
    __shared__ unsigned short WT[QKVW * 40];   // WT[col][kk], pitch 40
    int t = threadIdx.x;
    int lane = t & 31, wave = t >> 5, lane16 = lane & 15, half = lane >> 4;
    int r0 = blockIdx.x * 128 + wave * 16;
    int kp = t & 15, cg = t >> 4;              // fill mapping: kk pair x 20-col group

    v8f acc[20] = {};
    for (int k0 = 0; k0 < C3; k0 += 32) {
        __syncthreads();
        {   // cooperative transpose-fill of WT: 32 x 320 chunk, packed b32 stores
            union { v2u d[5]; unsigned short u[20]; } ra, rb;
            const unsigned short* s0 = wqkv + (size_t)(k0 + 2 * kp) * QKVW + cg * 20;
            #pragma unroll
            for (int i = 0; i < 5; ++i) ra.d[i] = *(const v2u*)(s0 + i * 4);
            #pragma unroll
            for (int i = 0; i < 5; ++i) rb.d[i] = *(const v2u*)(s0 + QKVW + i * 4);
            #pragma unroll
            for (int i = 0; i < 20; ++i)
                *(unsigned*)&WT[(cg * 20 + i) * 40 + 2 * kp] =
                    (unsigned)ra.u[i] | ((unsigned)rb.u[i] << 16);
        }
        __syncthreads();
        V16U a;   // A-operand: 16 rows of f, K-split {0-15 | 16-31}
        const unsigned short* frow = f + (size_t)(r0 + lane16) * C3 + k0;
        a.q[0] = *(const v4u*)(frow + half * 16);
        a.q[1] = *(const v4u*)(frow + half * 16 + 8);
        V16U bv[2];
        {
            const unsigned short* b0 = WT + lane16 * 40 + half * 16;
            bv[0].q[0] = *(const v4u*)(b0);
            bv[0].q[1] = *(const v4u*)(b0 + 8);
        }
        #pragma unroll
        for (int nt = 0; nt < 20; ++nt) {
            if (nt < 19) {   // prefetch next B tile before consuming current
                const unsigned short* bn = WT + ((nt + 1) * 16 + lane16) * 40 + half * 16;
                bv[(nt + 1) & 1].q[0] = *(const v4u*)(bn);
                bv[(nt + 1) & 1].q[1] = *(const v4u*)(bn + 8);
            }
            acc[nt] = __builtin_amdgcn_wmma_f32_16x16x32_bf16(
                false, a.v, false, bv[nt & 1].v, (short)0, acc[nt], false, false);
        }
    }
    #pragma unroll
    for (int nt = 0; nt < 20; ++nt)
        #pragma unroll
        for (int r = 0; r < 8; ++r) {
            int row = r0 + r + half * 8;                 // D layout: M = r (+8 for hi half)
            qkv[(size_t)row * QKVW + nt * 16 + lane16] = f2bf(acc[nt][r]);
        }
}

// ---------------------------------------------------------------------------
// Kernel 4: fused sigmoid-gated attention + residual.
//   For each 16-row query tile (1 wave): stream 32-key chunks:
//     S = Q @ K^T (2 WMMA) -> sigmoid -> bf16 via LDS transpose ->
//     acc += S @ V (16 WMMA, V chunk staged transposed in LDS, WG-shared,
//     B operands double-buffered).
//   Epilogue: out = gamma*acc + x.
// ---------------------------------------------------------------------------
__global__ __launch_bounds__(256) void pam_attention(const unsigned short* __restrict__ qkv,
                                                     const float* __restrict__ x,
                                                     const float* __restrict__ gammap,
                                                     float* __restrict__ out) {
    __shared__ unsigned short VT[256 * 40];    // VT[c][m], pitch 40 (WG-shared V chunk)
    __shared__ unsigned short SL[8 * 16 * 40]; // per-wave sigmoid(S) tile, pitch 40
    int t = threadIdx.x;
    int lane = t & 31, wave = t >> 5, lane16 = lane & 15, half = lane >> 4;
    int b = blockIdx.x >> 5;                   // 32 query blocks per batch
    int qblk = blockIdx.x & 31;
    int q0 = qblk * 128 + wave * 16;
    const unsigned short* qbase = qkv + (size_t)b * NN * QKVW;
    float gamma = gammap[0];

    V16U aq;   // whole Q tile: K-dim is 32 == one bf16 WMMA K
    {
        const unsigned short* qr = qbase + (size_t)(q0 + lane16) * QKVW;
        aq.q[0] = *(const v4u*)(qr + half * 16);
        aq.q[1] = *(const v4u*)(qr + half * 16 + 8);
    }
    v8f acc[16] = {};
    unsigned short* SW = SL + wave * 16 * 40;
    int mq = t >> 5;            // fill mapping: key quad (m = 4*mq)
    int cb = (t & 31) * 8;      // 8-channel group

    for (int m0 = 0; m0 < NN; m0 += 32) {
        __syncthreads();
        {   // transpose-fill VT: 4 V rows per thread -> packed ds_store_b64
            union { v4u q; unsigned short u[8]; } r0, r1, r2, r3;
            const unsigned short* s = qbase + (size_t)(m0 + 4 * mq) * QKVW + 64 + cb;
            r0.q = *(const v4u*)(s);
            r1.q = *(const v4u*)(s + QKVW);
            r2.q = *(const v4u*)(s + 2 * QKVW);
            r3.q = *(const v4u*)(s + 3 * QKVW);
            #pragma unroll
            for (int i = 0; i < 8; ++i) {
                v2u pk;
                pk.x = (unsigned)r0.u[i] | ((unsigned)r1.u[i] << 16);
                pk.y = (unsigned)r2.u[i] | ((unsigned)r3.u[i] << 16);
                *(v2u*)&VT[(cb + i) * 40 + 4 * mq] = pk;
            }
        }
        __syncthreads();
        // --- S = Q @ K^T : B-operand is K rows (row-contiguous in memory) ---
        v8f z = {};
        V16U bk0, bk1;
        const unsigned short* kr0 = qbase + (size_t)(m0 + lane16) * QKVW + QK + half * 16;
        bk0.q[0] = *(const v4u*)(kr0);
        bk0.q[1] = *(const v4u*)(kr0 + 8);
        const unsigned short* kr1 = qbase + (size_t)(m0 + 16 + lane16) * QKVW + QK + half * 16;
        bk1.q[0] = *(const v4u*)(kr1);
        bk1.q[1] = *(const v4u*)(kr1 + 8);
        v8f s1 = __builtin_amdgcn_wmma_f32_16x16x32_bf16(false, aq.v, false, bk0.v, (short)0, z, false, false);
        v8f s2 = __builtin_amdgcn_wmma_f32_16x16x32_bf16(false, aq.v, false, bk1.v, (short)0, z, false, false);
        // --- sigmoid, D-layout -> A-layout transpose through wave-private LDS ---
        #pragma unroll
        for (int r = 0; r < 8; ++r) {
            int M = r + half * 8;
            float p1 = __builtin_amdgcn_rcpf(1.f + __expf(-s1[r]));
            float p2 = __builtin_amdgcn_rcpf(1.f + __expf(-s2[r]));
            SW[M * 40 + lane16]      = f2bf(p1);
            SW[M * 40 + 16 + lane16] = f2bf(p2);
        }
        asm volatile("s_wait_dscnt 0x0" ::: "memory");   // wave-local LDS RAW ordering
        V16U as;
        const unsigned short* srow = SW + lane16 * 40;
        as.q[0] = *(const v4u*)(srow + half * 16);
        as.q[1] = *(const v4u*)(srow + half * 16 + 8);
        // --- acc += S @ V over 16 channel tiles, double-buffered B operands ---
        V16U bv[2];
        {
            const unsigned short* b0 = VT + lane16 * 40 + half * 16;
            bv[0].q[0] = *(const v4u*)(b0);
            bv[0].q[1] = *(const v4u*)(b0 + 8);
        }
        #pragma unroll
        for (int ct = 0; ct < 16; ++ct) {
            if (ct < 15) {   // prefetch next B tile before consuming current
                const unsigned short* bn = VT + ((ct + 1) * 16 + lane16) * 40 + half * 16;
                bv[(ct + 1) & 1].q[0] = *(const v4u*)(bn);
                bv[(ct + 1) & 1].q[1] = *(const v4u*)(bn + 8);
            }
            acc[ct] = __builtin_amdgcn_wmma_f32_16x16x32_bf16(
                false, as.v, false, bv[ct & 1].v, (short)0, acc[ct], false, false);
        }
    }
    // --- epilogue: out = gamma*acc + x ---
    const float* xb = x + (size_t)b * NN * CC;
    float* ob = out + (size_t)b * NN * CC;
    #pragma unroll
    for (int ct = 0; ct < 16; ++ct)
        #pragma unroll
        for (int r = 0; r < 8; ++r) {
            int row = q0 + r + half * 8;
            int c = ct * 16 + lane16;
            size_t idx = (size_t)row * CC + c;
            ob[idx] = gamma * acc[ct][r] + xb[idx];
        }
}

// ---------------------------------------------------------------------------
extern "C" void kernel_launch(void* const* d_in, const int* in_sizes, int n_in,
                              void* d_out, int out_size, void* d_ws, size_t ws_size,
                              hipStream_t stream) {
    const float* x     = (const float*)d_in[0];
    const float* wdw   = (const float*)d_in[1];
    const float* wp    = (const float*)d_in[2];
    const float* wb    = (const float*)d_in[3];
    const float* wc    = (const float*)d_in[4];
    const float* wd    = (const float*)d_in[5];
    const float* gamma = (const float*)d_in[6];

    char* ws = (char*)d_ws;
    unsigned short* fbuf = (unsigned short*)(ws);                         // 16384*768  bf16 = 25,165,824 B
    unsigned short* qkv  = (unsigned short*)(ws + 25165824);              // 16384*320  bf16 = 10,485,760 B
    unsigned short* wqkv = (unsigned short*)(ws + 25165824 + 10485760);   //   768*320  bf16 =    491,520 B

    pam_fuse_weights<<<(C3 * QKVW + 255) / 256, 256, 0, stream>>>(wp, wb, wc, wd, wqkv);
    pam_dwconv<<<(NPIX * C3 + 255) / 256, 256, 0, stream>>>(x, wdw, fbuf);
    pam_qkv_gemm<<<NPIX / 128, 256, 0, stream>>>(fbuf, wqkv, qkv);
    pam_attention<<<NBATCH * (NN / 128), 256, 0, stream>>>(qkv, x, gamma, (float*)d_out);
}
